// Sampler_40939628265869
// MI455X (gfx1250) — matrix operand order repaired and checked by reading, not compiled, not run on hardware
//
#include <hip/hip_runtime.h>
#include <hip/hip_bf16.h>
#include <stdint.h>

typedef __bf16 v16bf __attribute__((ext_vector_type(16)));
typedef __bf16 v8bf  __attribute__((ext_vector_type(8)));
typedef float  v8f   __attribute__((ext_vector_type(8)));

#define B_  64
#define L_  8
#define H_  4096
#define V_  50257
#define NT  ((V_ + 15) / 16)          // 3142 16-wide N tiles
#define WPB 4                          // waves per block

#define LOGITS_BYTES ((size_t)B_ * V_ * sizeof(float))   // 12,865,792 (256-aligned)

// ---------------------------------------------------------------------------
// Kernel 0: one-time f32 -> bf16 conversion of hs[:, pos, :]  (64 x 4096)
// ---------------------------------------------------------------------------
__global__ __launch_bounds__(256)
void cvt_hidden_bf16(const float* __restrict__ hs,
                     const int*   __restrict__ pos_p,
                     __bf16*      __restrict__ abuf)
{
  const int i = blockIdx.x * 256 + threadIdx.x;      // over 64*4096
  if (i < B_ * H_) {
    const int m = i >> 12;                            // / 4096
    const int k = i & (H_ - 1);
    abuf[i] = (__bf16)hs[((size_t)m * L_ + pos_p[0]) * H_ + k];
  }
}

// ---------------------------------------------------------------------------
// Kernel 1: logits = h @ emb^T + bias  via v_wmma_f32_16x16x32_bf16
// One wave computes a 64(M) x 16(N) tile; K=4096 in steps of 32,
// ping-pong software pipeline (two disjoint register sets, no rotation moves).
// ---------------------------------------------------------------------------
__device__ __forceinline__ v16bf cvtB(float4 a, float4 b, float4 c, float4 d) {
  v16bf r;
  r[0]=(__bf16)a.x;  r[1]=(__bf16)a.y;  r[2]=(__bf16)a.z;  r[3]=(__bf16)a.w;
  r[4]=(__bf16)b.x;  r[5]=(__bf16)b.y;  r[6]=(__bf16)b.z;  r[7]=(__bf16)b.w;
  r[8]=(__bf16)c.x;  r[9]=(__bf16)c.y;  r[10]=(__bf16)c.z; r[11]=(__bf16)c.w;
  r[12]=(__bf16)d.x; r[13]=(__bf16)d.y; r[14]=(__bf16)d.z; r[15]=(__bf16)d.w;
  return r;
}
__device__ __forceinline__ v16bf joinA(v8bf lo, v8bf hi) {
  return __builtin_shufflevector(lo, hi, 0,1,2,3,4,5,6,7,8,9,10,11,12,13,14,15);
}
__device__ __forceinline__ v8f wmma_bf16(v16bf a, v16bf b, v8f c) {
  return __builtin_amdgcn_wmma_f32_16x16x32_bf16(false, a, false, b,
                                                 (short)0, c, false, false);
}

// stage load: 4x b128 f32 (B stream) + 8x b128 bf16 (A fragments)
#define LOAD_STAGE(S, kk)                                                \
  S##_b0  = *(const float4*)(bptr + (kk) + 0);                           \
  S##_b1  = *(const float4*)(bptr + (kk) + 4);                           \
  S##_b2  = *(const float4*)(bptr + (kk) + 16);                          \
  S##_b3  = *(const float4*)(bptr + (kk) + 20);                          \
  S##_a0l = *(const v8bf*)(a0 + (kk));  S##_a0h = *(const v8bf*)(a0 + (kk) + 16); \
  S##_a1l = *(const v8bf*)(a1 + (kk));  S##_a1h = *(const v8bf*)(a1 + (kk) + 16); \
  S##_a2l = *(const v8bf*)(a2 + (kk));  S##_a2h = *(const v8bf*)(a2 + (kk) + 16); \
  S##_a3l = *(const v8bf*)(a3 + (kk));  S##_a3h = *(const v8bf*)(a3 + (kk) + 16)

#define CONSUME_STAGE(S)                                                 \
  do {                                                                   \
    const v16bf bf_ = cvtB(S##_b0, S##_b1, S##_b2, S##_b3);              \
    acc0 = wmma_bf16(joinA(S##_a0l, S##_a0h), bf_, acc0);                \
    acc1 = wmma_bf16(joinA(S##_a1l, S##_a1h), bf_, acc1);                \
    acc2 = wmma_bf16(joinA(S##_a2l, S##_a2h), bf_, acc2);                \
    acc3 = wmma_bf16(joinA(S##_a3l, S##_a3h), bf_, acc3);                \
  } while (0)

__global__ __launch_bounds__(32 * WPB)
void sampler_gemm_wmma(const float*  __restrict__ emb,
                       const __bf16* __restrict__ abuf,
                       const float*  __restrict__ bias,
                       float*        __restrict__ logits)
{
  const int lane = threadIdx.x & 31;
  const int wave = threadIdx.x >> 5;
  const int tile = blockIdx.x * WPB + wave;
  if (tile >= NT) return;

  const int col = lane & 15;
  const int hi  = lane >> 4;
  const int n   = tile * 16 + col;
  const int nc  = (n < V_) ? n : (V_ - 1);     // clamp; garbage cols never stored

  // B operand (emb^T): lane holds K = k0+8*hi+{0..7} and k0+16+8*hi+{0..7}, column n  (f32 stream)
  const float* bptr = emb + (size_t)nc * H_ + 8 * hi;
  // A operand (bf16): lane holds K = k0+8*hi+{0..7} and k0+16+8*hi+{0..7}
  const __bf16* a0 = abuf + (size_t)( 0 + col) * H_ + 8 * hi;
  const __bf16* a1 = abuf + (size_t)(16 + col) * H_ + 8 * hi;
  const __bf16* a2 = abuf + (size_t)(32 + col) * H_ + 8 * hi;
  const __bf16* a3 = abuf + (size_t)(48 + col) * H_ + 8 * hi;

  v8f acc0 = (v8f){0.f,0.f,0.f,0.f,0.f,0.f,0.f,0.f};
  v8f acc1 = acc0, acc2 = acc0, acc3 = acc0;

  // two disjoint pipeline stages (ping-pong)
  float4 X_b0, X_b1, X_b2, X_b3;
  v8bf   X_a0l, X_a0h, X_a1l, X_a1h, X_a2l, X_a2h, X_a3l, X_a3h;
  float4 Y_b0, Y_b1, Y_b2, Y_b3;
  v8bf   Y_a0l, Y_a0h, Y_a1l, Y_a1h, Y_a2l, Y_a2h, Y_a3l, Y_a3h;

  LOAD_STAGE(X, 0);

#pragma unroll 1
  for (int k0 = 0; k0 < H_; k0 += 64) {
    LOAD_STAGE(Y, k0 + 32);                          // always in range (<= 4064)
    __builtin_prefetch(bptr + k0 + 1024, 0, 3);      // ~4KB ahead on the B stream
    CONSUME_STAGE(X);

    const int kn = (k0 + 64 < H_) ? (k0 + 64) : 0;   // dummy wrap on last iter
    LOAD_STAGE(X, kn);
    CONSUME_STAGE(Y);
  }

  if (n < V_) {
    const float bn = bias[nc];
#pragma unroll
    for (int r = 0; r < 8; ++r) {
      const int m = r + 8 * hi;                      // C layout: vgpr r -> M=r / r+8
      logits[(size_t)(m +  0) * V_ + n] = acc0[r] + bn;
      logits[(size_t)(m + 16) * V_ + n] = acc1[r] + bn;
      logits[(size_t)(m + 32) * V_ + n] = acc2[r] + bn;
      logits[(size_t)(m + 48) * V_ + n] = acc3[r] + bn;
    }
  }
}

// ---------------------------------------------------------------------------
// Kernel 2: per-row softmax + exact top-64 radix-select + top-k/top-p +
// renormalize + deterministic categorical draw. One 256-thread block per row.
// ---------------------------------------------------------------------------
__device__ inline unsigned fkey(float f) {   // order-preserving float -> uint
  unsigned u = __float_as_uint(f);
  return (u & 0x80000000u) ? ~u : (u | 0x80000000u);
}

__global__ __launch_bounds__(256)
void sampler_select_sample(const float* __restrict__ logits,
                           const float* __restrict__ temps,
                           const float* __restrict__ topps,
                           const int*   __restrict__ topks,
                           float*       __restrict__ out)
{
  const int b   = blockIdx.x;
  const int tid = threadIdx.x;
  const float* lg = logits + (size_t)b * V_;
  const float invT = 1.0f / temps[b];

  __shared__ float    s_red[256];
  __shared__ unsigned s_hist[256];
  __shared__ unsigned s_prefix, s_mask;
  __shared__ int      s_need, s_cnt;
  __shared__ float    s_m, s_z;
  __shared__ float    s_cl[64];
  __shared__ int      s_ci[64];
  __shared__ float    s_pr[64];
  __shared__ int      s_kp[64];

  // ---- pass 1: row max of scaled logits
  float lmax = -INFINITY;
  for (int v = tid; v < V_; v += 256) lmax = fmaxf(lmax, lg[v] * invT);
  s_red[tid] = lmax; __syncthreads();
  for (int off = 128; off > 0; off >>= 1) {
    if (tid < off) s_red[tid] = fmaxf(s_red[tid], s_red[tid + off]);
    __syncthreads();
  }
  if (tid == 0) s_m = s_red[0];
  __syncthreads();
  const float m = s_m;

  // ---- pass 2: softmax denominator
  float lsum = 0.f;
  for (int v = tid; v < V_; v += 256) lsum += __expf(lg[v] * invT - m);
  s_red[tid] = lsum; __syncthreads();
  for (int off = 128; off > 0; off >>= 1) {
    if (tid < off) s_red[tid] += s_red[tid + off];
    __syncthreads();
  }
  if (tid == 0) s_z = s_red[0];
  __syncthreads();
  const float Z = s_z;

  // ---- radix select: exact key of the 64th-largest logit
  if (tid == 0) { s_prefix = 0u; s_mask = 0u; s_need = 64; }
  __syncthreads();
  for (int p = 3; p >= 0; --p) {
    s_hist[tid] = 0u;
    __syncthreads();
    const unsigned pref = s_prefix, msk = s_mask;
    for (int v = tid; v < V_; v += 256) {
      const unsigned k = fkey(lg[v]);
      if ((k & msk) == pref) atomicAdd(&s_hist[(k >> (8 * p)) & 255u], 1u);
    }
    __syncthreads();
    if (tid == 0) {
      unsigned cum = 0; const int need = s_need;
      for (int bin = 255; bin >= 0; --bin) {
        const unsigned c = s_hist[bin];
        if (cum + c >= (unsigned)need) {
          s_prefix = pref | ((unsigned)bin << (8 * p));
          s_mask   = msk  | (255u << (8 * p));
          s_need   = need - (int)cum;
          break;
        }
        cum += c;
      }
    }
    __syncthreads();
  }
  const unsigned thresh = s_prefix;

  // ---- collect top-64 (strictly-greater first, then ties up to 64)
  if (tid == 0) s_cnt = 0;
  __syncthreads();
  for (int v = tid; v < V_; v += 256) {
    if (fkey(lg[v]) > thresh) {
      const int i = atomicAdd(&s_cnt, 1);
      if (i < 64) { s_cl[i] = lg[v]; s_ci[i] = v; }
    }
  }
  __syncthreads();
  for (int v = tid; v < V_; v += 256) {
    if (fkey(lg[v]) == thresh) {
      const int i = atomicAdd(&s_cnt, 1);
      if (i < 64) { s_cl[i] = lg[v]; s_ci[i] = v; }
    }
  }
  __syncthreads();

  if (tid == 0) {
    // insertion sort: descending value, ties by ascending index (argsort order)
    for (int i = 1; i < 64; ++i) {
      const float lv = s_cl[i]; const int iv = s_ci[i];
      int j = i - 1;
      while (j >= 0 && (s_cl[j] < lv || (s_cl[j] == lv && s_ci[j] > iv))) {
        s_cl[j + 1] = s_cl[j]; s_ci[j + 1] = s_ci[j]; --j;
      }
      s_cl[j + 1] = lv; s_ci[j + 1] = iv;
    }
    // top-p / top-k mask + renormalization sum
    const int   kk = topks[b];
    const float tp = topps[b];
    float cum = 0.f, S = 0.f;
    for (int r = 0; r < 64; ++r) {
      const float pr = __expf(s_cl[r] * invT - m) / Z;
      const int keep = (r < kk) && (cum <= tp);   // mask: cum-before-self > top_p
      cum += pr;
      s_pr[r] = pr; s_kp[r] = keep;
      if (keep) S += pr;
    }
    // deterministic uniform draw (splitmix64 of (key=42, row b))
    uint64_t s = (42ull << 32) | (unsigned)b;
    s += 0x9E3779B97F4A7C15ull;
    s = (s ^ (s >> 30)) * 0xBF58476D1CE4E5B9ull;
    s = (s ^ (s >> 27)) * 0x94D049BB133111EBull;
    s ^= s >> 31;
    const float u = (float)(s >> 40) * (1.0f / 16777216.0f);
    const float target = u * S;
    float acc = 0.f; int chosen = s_ci[0];
    for (int r = 0; r < 64; ++r) {
      if (s_kp[r]) {
        chosen = s_ci[r];
        acc += s_pr[r];
        if (acc >= target) break;
      }
    }
    out[b] = (float)chosen;
  }
}

// ---------------------------------------------------------------------------
extern "C" void kernel_launch(void* const* d_in, const int* in_sizes, int n_in,
                              void* d_out, int out_size, void* d_ws, size_t ws_size,
                              hipStream_t stream) {
  const float* emb  = (const float*)d_in[0];   // (V, H) f32
  const float* hs   = (const float*)d_in[1];   // (B, L, H) f32
  const int*   pos  = (const int*)  d_in[2];   // scalar
  const float* temp = (const float*)d_in[3];   // (B,)
  const float* topp = (const float*)d_in[4];   // (B,)
  const int*   topk = (const int*)  d_in[5];   // (B,)
  const float* bias = (const float*)d_in[6];   // (V,)

  float*  logits = (float*)d_ws;                           // 12.9 MB
  __bf16* abuf   = (__bf16*)((char*)d_ws + LOGITS_BYTES);  // 512 KB bf16 A

  cvt_hidden_bf16<<<(B_ * H_ + 255) / 256, 256, 0, stream>>>(hs, pos, abuf);

  dim3 g1((NT + WPB - 1) / WPB);               // 786 blocks x 128 threads (4 waves)
  sampler_gemm_wmma<<<g1, 32 * WPB, 0, stream>>>(emb, abuf, bias, logits);

  sampler_select_sample<<<B_, 256, 0, stream>>>(logits, temp, topp, topk,
                                                (float*)d_out);
}
